// SparseFeatureGate_59176059404346
// MI455X (gfx1250) — compile-verified, matrix operand order checked by reading, not run on hardware
//
#include <hip/hip_runtime.h>
#include <hip/hip_bf16.h>

// ---------------------------------------------------------------------------
// Problem constants (match reference setup_inputs)
// ---------------------------------------------------------------------------
static constexpr int Bc = 32;
static constexpr int Tc = 512;
static constexpr int Ec = 512;
static constexpr int Hc = 8;
static constexpr int Nc = Bc * Tc;          // 16384 rows
static constexpr float GAMMA_C = 1.5f;
static constexpr float BN_EPS_C = 1e-5f;

typedef __attribute__((ext_vector_type(16))) __bf16 v16bf;
typedef __attribute__((ext_vector_type(8)))  __bf16 v8bf;
typedef __attribute__((ext_vector_type(8)))  float  v8f;
typedef __attribute__((ext_vector_type(4)))  float  f32x4;

// ---------------------------------------------------------------------------
// bf16 <-> f32 helpers (RNE on the way down)
// ---------------------------------------------------------------------------
__device__ __forceinline__ __bf16 f2bf(float f) {
    unsigned u = __builtin_bit_cast(unsigned, f);
    u += 0x7FFFu + ((u >> 16) & 1u);
    unsigned short hs = (unsigned short)(u >> 16);
    return __builtin_bit_cast(__bf16, hs);
}
__device__ __forceinline__ float bf2f(__bf16 b) {
    unsigned short hs = __builtin_bit_cast(unsigned short, b);
    unsigned u = ((unsigned)hs) << 16;
    return __builtin_bit_cast(float, u);
}

// ---------------------------------------------------------------------------
// CDNA5 async global->LDS copy (16 bytes / lane), tracked by ASYNCcnt
// ---------------------------------------------------------------------------
__device__ __forceinline__ void async_cp16(unsigned lds_addr, const __bf16* gptr) {
    asm volatile("global_load_async_to_lds_b128 %0, %1, off"
                 :: "v"(lds_addr), "v"((unsigned long long)(size_t)gptr)
                 : "memory");
}
__device__ __forceinline__ void wait_async0() {
    asm volatile("s_wait_asynccnt 0x0" ::: "memory");
}
__device__ __forceinline__ unsigned lds_off(const void* p) {
    // low 32 bits of a generic pointer to __shared__ == LDS byte address
    return (unsigned)(size_t)p;
}

// ---------------------------------------------------------------------------
// Kernel 0a: one-time fp32 -> bf16 conversion (8 elements / thread)
// ---------------------------------------------------------------------------
__global__ void cvt_f32_to_bf16(const float* __restrict__ in,
                                __bf16* __restrict__ out, int nvec8)
{
    int i = blockIdx.x * blockDim.x + threadIdx.x;
    if (i < nvec8) {
        const size_t base = (size_t)i * 8;
        f32x4 a = *(const f32x4*)(in + base);
        f32x4 b = *(const f32x4*)(in + base + 4);
        v8bf o;
#pragma unroll
        for (int j = 0; j < 4; ++j) { o[j] = f2bf(a[j]); o[4 + j] = f2bf(b[j]); }
        *(v8bf*)(out + base) = o;
    }
}

// ---------------------------------------------------------------------------
// Kernel 0b: zero the BN stat accumulators
// ---------------------------------------------------------------------------
__global__ void init_stats(float* __restrict__ sum, float* __restrict__ sumsq) {
    int i = blockIdx.x * blockDim.x + threadIdx.x;
    if (i < Hc * Ec) { sum[i] = 0.0f; sumsq[i] = 0.0f; }
}

// ---------------------------------------------------------------------------
// bf16 WMMA fragment gather from LDS: 16 bf16 per lane, two runs of 8
// (ISA 7.12.2 bf16 16x32 layout: slots 0..7 <- K+{0..7}, slots 8..15 <- K+16+{0..7},
//  K pre-offset by 8*half for lanes 16..31). 16-byte aligned ds_load_b128 pairs.
// ---------------------------------------------------------------------------
__device__ __forceinline__ v16bf load_frag_lds(const __bf16* p) {
    v8bf lo = *(const v8bf*)(p);
    v8bf hi = *(const v8bf*)(p + 16);
    return __builtin_shufflevector(lo, hi, 0, 1, 2, 3, 4, 5, 6, 7,
                                   8, 9, 10, 11, 12, 13, 14, 15);
}

// ---------------------------------------------------------------------------
// Kernel 1: batched GEMM  logits[h] = X * W[h]^T  via bf16 WMMA with
// double-buffered async global->LDS staging.
// Block: 256 threads = 8 waves (2 M x 4 N), block tile 64(M) x 128(N).
// Per K-step the block stages A(64x32) + B(128x32) into LDS once (async),
// each wave computes a 32x32 C region: 2 A-frags x 2 B-frags -> 4 WMMAs.
// Fused BN-statistics epilogue (f32 atomics).
// Grid: (E/128, N/64, H) = (4, 256, 8)
// ---------------------------------------------------------------------------
static constexpr int LSTR = 40;   // padded LDS row stride (80 B, 16B-aligned)

__global__ __launch_bounds__(256)
void gemm_bn_stats(const __bf16* __restrict__ xb,   // (N, E) bf16 row-major
                   const __bf16* __restrict__ Wb,   // (H, E, E) bf16: W[h][o][i]
                   __bf16* __restrict__ logits,     // (H, N, E) bf16
                   float* __restrict__ sum,         // (H, E)
                   float* __restrict__ sumsq)       // (H, E)
{
    __shared__ alignas(16) __bf16 As[2][64 * LSTR];    // 10.0 KB
    __shared__ alignas(16) __bf16 Bs[2][128 * LSTR];   // 20.0 KB

    const int h    = blockIdx.z;
    const int tid  = threadIdx.x;
    const int lane = tid & 31;
    const int wave = tid >> 5;
    const int half = lane >> 4;      // 0: lanes 0-15, 1: lanes 16-31
    const int mrow = lane & 15;
    const int wm   = wave >> 2;      // 0..1
    const int wn   = wave & 3;       // 0..3

    const int gm = blockIdx.y * 64;      // block row base
    const int gn = blockIdx.x * 128;     // block col base

    // staging assignment: A = 64 rows x 4 chunks (1/thread), B = 128 rows x 4 (2/thread)
    const int ar  = tid >> 2;                  // A row 0..63
    const int ac  = (tid & 3) * 8;             // A col chunk
    const int br0 = tid >> 1;                  // B rows for chunk pair
    const int bc0 = (tid & 1) * 16;            // two 8-elem chunks at +0,+8

    const __bf16* agp = xb + (size_t)(gm + ar) * Ec + ac;
    const __bf16* bgp = Wb + ((size_t)h * Ec + (size_t)(gn + br0)) * Ec + bc0;

    const unsigned aLds[2] = { lds_off(&As[0][ar * LSTR + ac]),
                               lds_off(&As[1][ar * LSTR + ac]) };
    const unsigned bLds[2] = { lds_off(&Bs[0][br0 * LSTR + bc0]),
                               lds_off(&Bs[1][br0 * LSTR + bc0]) };

    // compute-side LDS fragment bases (per wave/lane)
    const __bf16* a0p[2] = { &As[0][(wm * 32 + mrow) * LSTR + 8 * half],
                             &As[1][(wm * 32 + mrow) * LSTR + 8 * half] };
    const __bf16* a1p[2] = { &As[0][(wm * 32 + 16 + mrow) * LSTR + 8 * half],
                             &As[1][(wm * 32 + 16 + mrow) * LSTR + 8 * half] };
    const __bf16* b0p[2] = { &Bs[0][(wn * 32 + mrow) * LSTR + 8 * half],
                             &Bs[1][(wn * 32 + mrow) * LSTR + 8 * half] };
    const __bf16* b1p[2] = { &Bs[0][(wn * 32 + 16 + mrow) * LSTR + 8 * half],
                             &Bs[1][(wn * 32 + 16 + mrow) * LSTR + 8 * half] };

    v8f c00 = {}, c01 = {}, c10 = {}, c11 = {};

    // prologue: stage K-slice 0 into buffer 0
    async_cp16(aLds[0], agp);
    async_cp16(bLds[0], bgp);
    async_cp16(bLds[0] + 8 * 2, bgp + 8);
    wait_async0();
    __syncthreads();

    int cur = 0;
    for (int k0 = 0; k0 < Ec; k0 += 32) {
        // stage next K-slice into the other buffer while computing on this one
        if (k0 + 32 < Ec) {
            const int nxt = cur ^ 1;
            async_cp16(aLds[nxt], agp + (k0 + 32));
            async_cp16(bLds[nxt], bgp + (k0 + 32));
            async_cp16(bLds[nxt] + 8 * 2, bgp + (k0 + 32) + 8);
        }

        v16bf a0 = load_frag_lds(a0p[cur]);
        v16bf a1 = load_frag_lds(a1p[cur]);
        v16bf b0 = load_frag_lds(b0p[cur]);
        v16bf b1 = load_frag_lds(b1p[cur]);

        c00 = __builtin_amdgcn_wmma_f32_16x16x32_bf16(false, a0, false, b0,
                                                      (short)0, c00, false, false);
        c01 = __builtin_amdgcn_wmma_f32_16x16x32_bf16(false, a0, false, b1,
                                                      (short)0, c01, false, false);
        c10 = __builtin_amdgcn_wmma_f32_16x16x32_bf16(false, a1, false, b0,
                                                      (short)0, c10, false, false);
        c11 = __builtin_amdgcn_wmma_f32_16x16x32_bf16(false, a1, false, b1,
                                                      (short)0, c11, false, false);

        // my async copies done; barrier closes RAW (all copies) + WAR (all reads)
        wait_async0();
        __syncthreads();
        cur ^= 1;
    }

    // Epilogue: C layout per acc: VGPR r -> M = mbase + r + 8*half, N = nbase + mrow.
    // Store bf16 logits; fold per-column sums across both M sub-tiles and the
    // half-wave pair, then one atomic per (stat, column).
    const int m0 = gm + wm * 32;
    const int n0 = gn + wn * 32;
#pragma unroll
    for (int nt = 0; nt < 2; ++nt) {
        const int ncol = n0 + nt * 16 + mrow;
        float s = 0.0f, ss = 0.0f;
#pragma unroll
        for (int mt = 0; mt < 2; ++mt) {
            const v8f c = (mt == 0) ? (nt == 0 ? c00 : c01)
                                    : (nt == 0 ? c10 : c11);
            const int mbase = m0 + mt * 16 + 8 * half;
#pragma unroll
            for (int r = 0; r < 8; ++r) {
                const float v = c[r];
                logits[((size_t)h * Nc + (mbase + r)) * Ec + ncol] = f2bf(v);
                s  += v;
                ss += v * v;
            }
        }
        s  += __shfl_xor(s, 16, 32);
        ss += __shfl_xor(ss, 16, 32);
        if (half == 0) {
            atomicAdd(&sum[h * Ec + ncol], s);
            atomicAdd(&sumsq[h * Ec + ncol], ss);
        }
    }
}

// ---------------------------------------------------------------------------
// Kernel 2: finalize BN stats -> mean, rstd
// ---------------------------------------------------------------------------
__global__ void finalize_stats(const float* __restrict__ sum,
                               const float* __restrict__ sumsq,
                               float* __restrict__ mean,
                               float* __restrict__ rstd)
{
    int i = blockIdx.x * blockDim.x + threadIdx.x;
    if (i < Hc * Ec) {
        float m = sum[i] * (1.0f / (float)Nc);
        float v = sumsq[i] * (1.0f / (float)Nc) - m * m;
        mean[i] = m;
        rstd[i] = rsqrtf(fmaxf(v, 0.0f) + BN_EPS_C);
    }
}

// ---------------------------------------------------------------------------
// Block reductions (wave32 shuffles + 8-slot LDS)
// ---------------------------------------------------------------------------
__device__ __forceinline__ float wave_sum(float v) {
#pragma unroll
    for (int off = 16; off > 0; off >>= 1) v += __shfl_down(v, off, 32);
    return v;
}
__device__ __forceinline__ float wave_max(float v) {
#pragma unroll
    for (int off = 16; off > 0; off >>= 1) v = fmaxf(v, __shfl_down(v, off, 32));
    return v;
}
__device__ __forceinline__ float block_sum(float v, float* red) {
    v = wave_sum(v);
    __syncthreads();
    if ((threadIdx.x & 31) == 0) red[threadIdx.x >> 5] = v;
    __syncthreads();
    if (threadIdx.x == 0) {
        float t = 0.0f;
#pragma unroll
        for (int i = 0; i < 8; ++i) t += red[i];
        red[8] = t;
    }
    __syncthreads();
    return red[8];
}
__device__ __forceinline__ float block_max(float v, float* red) {
    v = wave_max(v);
    __syncthreads();
    if ((threadIdx.x & 31) == 0) red[threadIdx.x >> 5] = v;
    __syncthreads();
    if (threadIdx.x == 0) {
        float t = red[0];
#pragma unroll
        for (int i = 1; i < 8; ++i) t = fmaxf(t, red[i]);
        red[8] = t;
    }
    __syncthreads();
    return red[8];
}

// ---------------------------------------------------------------------------
// Kernel 3: per-row sequential-over-heads sparsemax gate.
// One block per row n, 256 threads, 2 columns/thread (E=512).
// Sparsemax tau via exact Michelot iteration (no sort needed).
// ---------------------------------------------------------------------------
__global__ __launch_bounds__(256)
void gate_kernel(const float* __restrict__ x,       // (N, E)
                 const __bf16* __restrict__ logits, // (H, N, E)
                 const float* __restrict__ mean,    // (H, E)
                 const float* __restrict__ rstd,    // (H, E)
                 const float* __restrict__ bnw,     // (H, E)
                 const float* __restrict__ bnb,     // (H, E)
                 float* __restrict__ out)           // masked_xs (H,B,T,E) ++ masks (B,H,T,E)
{
    const int n   = blockIdx.x;
    const int tid = threadIdx.x;
    const int e0  = tid;
    const int e1  = tid + 256;
    const int b   = n / Tc;
    const int t   = n - b * Tc;

    __shared__ float red[9];

    const float x0 = x[(size_t)n * Ec + e0];
    const float x1 = x[(size_t)n * Ec + e1];
    float p0 = 1.0f, p1 = 1.0f;

    float* out_masked = out;                              // (H, N, E)
    float* out_masks  = out + (size_t)Hc * Nc * Ec;       // (B, H, T, E)

    for (int h = 0; h < Hc; ++h) {
        const size_t li = ((size_t)h * Nc + n) * Ec;
        const int s0 = h * Ec + e0;
        const int s1 = h * Ec + e1;

        float z0 = (bf2f(logits[li + e0]) - mean[s0]) * rstd[s0] * bnw[s0] + bnb[s0];
        float z1 = (bf2f(logits[li + e1]) - mean[s1]) * rstd[s1] * bnw[s1] + bnb[s1];
        z0 *= p0;
        z1 *= p1;

        // numerical shift (sparsemax is shift-invariant)
        const float mx = block_max(fmaxf(z0, z1), red);
        z0 -= mx;
        z1 -= mx;

        // Michelot projection onto the simplex: refine tau until support stable
        bool a0 = true, a1 = true;
        float tau = 0.0f;
        for (int it = 0; it < 64; ++it) {
            const float S = block_sum((a0 ? z0 : 0.0f) + (a1 ? z1 : 0.0f), red);
            const float C = block_sum((a0 ? 1.0f : 0.0f) + (a1 ? 1.0f : 0.0f), red);
            tau = (S - 1.0f) / fmaxf(C, 1.0f);
            const bool na0 = z0 > tau;
            const bool na1 = z1 > tau;
            const int changed = (na0 != a0) | (na1 != a1);
            a0 = na0;
            a1 = na1;
            if (!__syncthreads_or(changed)) break;
        }

        const float m0v = fmaxf(z0 - tau, 0.0f);
        const float m1v = fmaxf(z1 - tau, 0.0f);

        // prior update: gamma - mask in [0.5, 1.5] so prior stays positive
        p0 *= fmaxf(GAMMA_C - m0v, 0.0f);
        p1 *= fmaxf(GAMMA_C - m1v, 0.0f);

        out_masked[li + e0] = x0 * m0v;
        out_masked[li + e1] = x1 * m1v;
        const size_t mi = (((size_t)b * Hc + h) * Tc + t) * (size_t)Ec;
        out_masks[mi + e0] = m0v;
        out_masks[mi + e1] = m1v;
    }
}

// ---------------------------------------------------------------------------
// Launch
// ---------------------------------------------------------------------------
extern "C" void kernel_launch(void* const* d_in, const int* in_sizes, int n_in,
                              void* d_out, int out_size, void* d_ws, size_t ws_size,
                              hipStream_t stream)
{
    const float* x   = (const float*)d_in[0];   // (B,T,E)
    const float* W   = (const float*)d_in[1];   // (H,E,E)
    const float* bnw = (const float*)d_in[2];   // (H,E)
    const float* bnb = (const float*)d_in[3];   // (H,E)
    float* out = (float*)d_out;

    // workspace layout:
    //   logits bf16 (H*N*E = 128 MB) | xb bf16 (16 MB) | Wb bf16 (4 MB) | stats (64 KB)
    char* ws = (char*)d_ws;
    __bf16* logits = (__bf16*)ws;
    size_t off = (size_t)Hc * Nc * Ec * sizeof(__bf16);
    __bf16* xb = (__bf16*)(ws + off); off += (size_t)Nc * Ec * sizeof(__bf16);
    __bf16* Wb = (__bf16*)(ws + off); off += (size_t)Hc * Ec * Ec * sizeof(__bf16);
    float* sum   = (float*)(ws + off); off += (size_t)Hc * Ec * sizeof(float);
    float* sumsq = (float*)(ws + off); off += (size_t)Hc * Ec * sizeof(float);
    float* mean  = (float*)(ws + off); off += (size_t)Hc * Ec * sizeof(float);
    float* rstd  = (float*)(ws + off);

    (void)in_sizes; (void)n_in; (void)out_size; (void)ws_size;

    // one-time bf16 conversion of both GEMM operands
    const int xv = (Nc * Ec) / 8;            // 1,048,576 vec8
    const int wv = (Hc * Ec * Ec) / 8;       //   262,144 vec8
    cvt_f32_to_bf16<<<(xv + 255) / 256, 256, 0, stream>>>(x, xb, xv);
    cvt_f32_to_bf16<<<(wv + 255) / 256, 256, 0, stream>>>(W, Wb, wv);

    init_stats<<<(Hc * Ec + 255) / 256, 256, 0, stream>>>(sum, sumsq);

    dim3 g(Ec / 128, Nc / 64, Hc);   // (4, 256, 8)
    gemm_bn_stats<<<g, 256, 0, stream>>>(xb, Wb, logits, sum, sumsq);

    finalize_stats<<<(Hc * Ec + 255) / 256, 256, 0, stream>>>(sum, sumsq, mean, rstd);

    gate_kernel<<<Nc, 256, 0, stream>>>(x, logits, mean, rstd, bnw, bnb, out);
}
